// S5SSM_52304111730983
// MI455X (gfx1250) — matrix operand order repaired and checked by reading, not compiled, not run on hardware
//
#include <hip/hip_runtime.h>
#include <hip/hip_bf16.h>

// S5 SSM forward for MI455X (gfx1250, wave32).
// Pipeline: prep -> WMMA bf16 GEMM (Bu) -> 3-phase reset-aware scan -> WMMA bf16 GEMM (ys).

#define L_SEQ 32768
#define H_DIM 256
#define P_DIM 256
#define CH    128                 // scan chunk length
#define NCH   (L_SEQ / CH)        // 256 chunks

typedef __attribute__((ext_vector_type(16))) __bf16 v16bf;
typedef __attribute__((ext_vector_type(8)))  float  v8f;

union BFrag {
  v16bf v;
  unsigned short h[16];
  uint4 q[2];
};

__device__ __forceinline__ unsigned short f2bf(float f) {
  unsigned int u = __float_as_uint(f);
  u += 0x7FFFu + ((u >> 16) & 1u);          // round-to-nearest-even
  return (unsigned short)(u >> 16);
}

// ---- workspace byte offsets ------------------------------------------------
#define WS_LB_RE   0u                        // P f32  : Lambda_bar.re
#define WS_LB_IM   1024u                     // P f32  : Lambda_bar.im
#define WS_F_RE    2048u                     // P f32  : ((Lb-1)/Lam).re
#define WS_F_IM    3072u                     // P f32  : ((Lb-1)/Lam).im
#define WS_BBT     4096u                     // 512x256 bf16 (B_bar packed, transposed)
#define WS_C2T     (WS_BBT + 262144u)        // 256x512 bf16 (2*Cr | -2*Ci, transposed)
#define WS_AER     (WS_C2T + 262144u)        // NCH*P f32 chunk A_eff.re
#define WS_AEI     (WS_AER + 262144u)
#define WS_CER     (WS_AEI + 262144u)        // chunk c_eff.re
#define WS_CEI     (WS_CER + 262144u)
#define WS_XR      (WS_CEI + 262144u)        // chunk-entry state re
#define WS_XI      (WS_XR  + 262144u)
#define WS_BU      (WS_XI  + 262144u)        // L x 512 f32 : [Bu_re | Bu_im], becomes xs

// ---------------------------------------------------------------------------
// Prep: per-mode discretization. Lambda_bar = exp(Lam*step); F = (Lb-1)/Lam.
__global__ void __launch_bounds__(256)
prep_modes(const float* __restrict__ lre, const float* __restrict__ lim,
           const float* __restrict__ logstep,
           float* __restrict__ lbr, float* __restrict__ lbi,
           float* __restrict__ fr,  float* __restrict__ fi) {
  const int p = threadIdx.x;
  const float step = __expf(logstep[p]);
  const float lr = lre[p], li = lim[p];
  const float er = __expf(lr * step);
  const float Lbr = er * __cosf(li * step);
  const float Lbi = er * __sinf(li * step);
  lbr[p] = Lbr; lbi[p] = Lbi;
  const float den = lr * lr + li * li;
  fr[p] = ((Lbr - 1.f) * lr + Lbi * li) / den;
  fi[p] = (Lbi * lr - (Lbr - 1.f) * li) / den;
}

// Bbt[j][h] (j<256: Re(B_bar[p=j]), j>=256: Im(B_bar[p=j-256])), bf16, pitch H.
__global__ void __launch_bounds__(256)
prep_bbar(const float* __restrict__ B, const float* __restrict__ fr,
          const float* __restrict__ fi, unsigned short* __restrict__ bbt) {
  const int p = blockIdx.x, h = threadIdx.x;
  const float br = B[((size_t)p * H_DIM + h) * 2 + 0];
  const float bi = B[((size_t)p * H_DIM + h) * 2 + 1];
  const float Fr = fr[p], Fi = fi[p];
  bbt[(size_t)p * H_DIM + h]             = f2bf(Fr * br - Fi * bi);
  bbt[(size_t)(P_DIM + p) * H_DIM + h]   = f2bf(Fr * bi + Fi * br);
}

// C2t[h][k] (k<256: 2*Cr[h][k], k>=256: -2*Ci[h][k-256]), bf16, pitch 512.
__global__ void __launch_bounds__(256)
prep_c2t(const float* __restrict__ C, unsigned short* __restrict__ c2t) {
  const int h = blockIdx.x, p = threadIdx.x;
  const float cr = C[((size_t)h * P_DIM + p) * 2 + 0];
  const float ci = C[((size_t)h * P_DIM + p) * 2 + 1];
  c2t[(size_t)h * 512 + p]       = f2bf(2.f * cr);
  c2t[(size_t)h * 512 + 256 + p] = f2bf(-2.f * ci);
}

// ---------------------------------------------------------------------------
// Shared GEMM inner step: load A run (f32 -> bf16 via hardware cvt), preload all
// four B fragments into distinct registers (enables partial s_wait_loadcnt
// overlap), then issue four back-to-back WMMAs.
__device__ __forceinline__ void
gemm_step(const float* __restrict__ a1, const unsigned short* __restrict__ bbase,
          size_t bpitch, int coloff, v8f acc[4]) {
  float tmp[16];
  *(float4*)(tmp + 0)  = *(const float4*)(a1 + 0);
  *(float4*)(tmp + 4)  = *(const float4*)(a1 + 4);
  *(float4*)(tmp + 8)  = *(const float4*)(a1 + 8);
  *(float4*)(tmp + 12) = *(const float4*)(a1 + 12);
  v16bf av;
  #pragma unroll
  for (int j = 0; j < 16; ++j) av[j] = (__bf16)tmp[j];   // v_cvt_*_bf16_f32

  BFrag b0, b1, b2, b3;
  {
    const unsigned short* bp = bbase + (size_t)(coloff + 0)  * bpitch;
    b0.q[0] = *(const uint4*)(bp + 0); b0.q[1] = *(const uint4*)(bp + 8);
  }
  {
    const unsigned short* bp = bbase + (size_t)(coloff + 16) * bpitch;
    b1.q[0] = *(const uint4*)(bp + 0); b1.q[1] = *(const uint4*)(bp + 8);
  }
  {
    const unsigned short* bp = bbase + (size_t)(coloff + 32) * bpitch;
    b2.q[0] = *(const uint4*)(bp + 0); b2.q[1] = *(const uint4*)(bp + 8);
  }
  {
    const unsigned short* bp = bbase + (size_t)(coloff + 48) * bpitch;
    b3.q[0] = *(const uint4*)(bp + 0); b3.q[1] = *(const uint4*)(bp + 8);
  }
  acc[0] = __builtin_amdgcn_wmma_f32_16x16x32_bf16(false, av, false, b0.v, (short)0, acc[0], false, false);
  acc[1] = __builtin_amdgcn_wmma_f32_16x16x32_bf16(false, av, false, b1.v, (short)0, acc[1], false, false);
  acc[2] = __builtin_amdgcn_wmma_f32_16x16x32_bf16(false, av, false, b2.v, (short)0, acc[2], false, false);
  acc[3] = __builtin_amdgcn_wmma_f32_16x16x32_bf16(false, av, false, b3.v, (short)0, acc[3], false, false);
}

// GEMM1: Bu[L x 512] = u[L x 256] (f32 -> bf16) @ Bbt^T. Wave = 16x64 strip.
__global__ void __launch_bounds__(256)
gemm_bu(const float* __restrict__ u, const unsigned short* __restrict__ bbt,
        float* __restrict__ bu) {
  const int w    = threadIdx.x >> 5;
  const int lane = threadIdx.x & 31;
  const int hi   = lane >> 4;      // K-half selector per ISA layout
  const int m    = lane & 15;      // A row / B,C column within tile
  const int rowt = blockIdx.x * 128 + w * 16;
  const int colb = blockIdx.y * 64;

  v8f acc[4];
  #pragma unroll
  for (int t = 0; t < 4; ++t)
    #pragma unroll
    for (int r = 0; r < 8; ++r) acc[t][r] = 0.0f;

  const float* arow = u + (size_t)(rowt + m) * H_DIM;
  for (int k = 0; k < H_DIM; k += 32) {
    const float* a1 = arow + k + hi * 16;
    if (k + 32 < H_DIM) __builtin_prefetch(a1 + 32, 0, 0);   // global_prefetch_b8
    // B fragment: lane holds column n, halves j -> K = hi*16 + j (contiguous in Bbt)
    gemm_step(a1, bbt + (size_t)(colb + m) * H_DIM + k + hi * 16, H_DIM, 0, acc);
  }

  #pragma unroll
  for (int t = 0; t < 4; ++t) {
    const int col = colb + t * 16 + m;
    float* orow = bu + (size_t)(rowt + hi * 8) * 512 + col;
    #pragma unroll
    for (int r = 0; r < 8; ++r) orow[(size_t)r * 512] = acc[t][r];
  }
}

// ---------------------------------------------------------------------------
// Scan phase 1: per (chunk, p) affine summary with resets.
__global__ void __launch_bounds__(256)
scan_chunk_reduce(const float* __restrict__ bu, const unsigned char* __restrict__ start,
                  const float* __restrict__ lbr, const float* __restrict__ lbi,
                  float* __restrict__ aer, float* __restrict__ aei,
                  float* __restrict__ cer, float* __restrict__ cei) {
  __shared__ unsigned char ss[CH];
  const int c = blockIdx.x, p = threadIdx.x;
  if (p < CH) ss[p] = start[(size_t)c * CH + p];
  __syncthreads();
  const float ar = lbr[p], ai = lbi[p];
  float Ar = 1.f, Ai = 0.f, Cr = 0.f, Ci = 0.f;
  const float* b = bu + (size_t)c * CH * 512;
  for (int i = 0; i < CH; ++i) {
    const float br = b[(size_t)i * 512 + p];
    const float bi = b[(size_t)i * 512 + 256 + p];
    if (ss[i]) { Ar = 0.f; Ai = 0.f; Cr = br; Ci = bi; }
    else {
      float nr = ar * Cr - ai * Ci + br;
      Ci = ar * Ci + ai * Cr + bi; Cr = nr;
      nr = ar * Ar - ai * Ai;
      Ai = ar * Ai + ai * Ar; Ar = nr;
    }
  }
  const size_t idx = (size_t)c * P_DIM + p;
  aer[idx] = Ar; aei[idx] = Ai; cer[idx] = Cr; cei[idx] = Ci;
}

// Scan phase 2: single-block carry sweep across chunks.
__global__ void __launch_bounds__(256)
scan_carry(const float* __restrict__ aer, const float* __restrict__ aei,
           const float* __restrict__ cer, const float* __restrict__ cei,
           const float* __restrict__ hidden,
           float* __restrict__ xr, float* __restrict__ xi) {
  const int p = threadIdx.x;
  float r = hidden[p], im = 0.f;
  for (int c = 0; c < NCH; ++c) {
    const size_t idx = (size_t)c * P_DIM + p;
    xr[idx] = r; xi[idx] = im;
    const float Ar = aer[idx], Ai = aei[idx], Cr = cer[idx], Ci = cei[idx];
    const float nr = Ar * r - Ai * im + Cr;
    im = Ar * im + Ai * r + Ci; r = nr;
  }
}

// Scan phase 3: replay chunks from entry state; overwrite Bu with xs; emit final state.
__global__ void __launch_bounds__(256)
scan_apply(float* __restrict__ bu, const unsigned char* __restrict__ start,
           const float* __restrict__ lbr, const float* __restrict__ lbi,
           const float* __restrict__ xr0, const float* __restrict__ xi0,
           float* __restrict__ out_final) {
  __shared__ unsigned char ss[CH];
  const int c = blockIdx.x, p = threadIdx.x;
  if (p < CH) ss[p] = start[(size_t)c * CH + p];
  __syncthreads();
  const float ar = lbr[p], ai = lbi[p];
  const size_t idx = (size_t)c * P_DIM + p;
  float r = xr0[idx], im = xi0[idx];
  float* b = bu + (size_t)c * CH * 512;
  for (int i = 0; i < CH; ++i) {
    const float br = b[(size_t)i * 512 + p];
    const float bi = b[(size_t)i * 512 + 256 + p];
    if (ss[i]) { r = br; im = bi; }
    else { const float nr = ar * r - ai * im + br; im = ar * im + ai * r + bi; r = nr; }
    b[(size_t)i * 512 + p]       = r;
    b[(size_t)i * 512 + 256 + p] = im;
  }
  if (c == NCH - 1) { out_final[2 * p] = r; out_final[2 * p + 1] = im; }  // complex64 layout
}

// ---------------------------------------------------------------------------
// GEMM2: ys[L x 256] = xs[L x 512] (f32 -> bf16) @ C2t^T + u*D.
__global__ void __launch_bounds__(256)
gemm_ys(const float* __restrict__ xs, const unsigned short* __restrict__ c2t,
        const float* __restrict__ u, const float* __restrict__ D,
        float* __restrict__ ys) {
  const int w    = threadIdx.x >> 5;
  const int lane = threadIdx.x & 31;
  const int hi   = lane >> 4;
  const int m    = lane & 15;
  const int rowt = blockIdx.x * 128 + w * 16;
  const int colb = blockIdx.y * 64;

  v8f acc[4];
  #pragma unroll
  for (int t = 0; t < 4; ++t)
    #pragma unroll
    for (int r = 0; r < 8; ++r) acc[t][r] = 0.0f;

  const float* arow = xs + (size_t)(rowt + m) * 512;
  for (int k = 0; k < 512; k += 32) {
    const float* a1 = arow + k + hi * 16;
    if (k + 32 < 512) __builtin_prefetch(a1 + 32, 0, 0);
    gemm_step(a1, c2t + (size_t)(colb + m) * 512 + k + hi * 16, 512, 0, acc);
  }

  #pragma unroll
  for (int t = 0; t < 4; ++t) {
    const int col = colb + t * 16 + m;
    const float d = D[col];
    #pragma unroll
    for (int r = 0; r < 8; ++r) {
      const int row = rowt + hi * 8 + r;
      ys[(size_t)row * H_DIM + col] = acc[t][r] + u[(size_t)row * H_DIM + col] * d;
    }
  }
}

// ---------------------------------------------------------------------------
extern "C" void kernel_launch(void* const* d_in, const int* in_sizes, int n_in,
                              void* d_out, int out_size, void* d_ws, size_t ws_size,
                              hipStream_t stream) {
  (void)in_sizes; (void)n_in; (void)out_size; (void)ws_size;
  const float* Lambda_re = (const float*)d_in[0];
  const float* Lambda_im = (const float*)d_in[1];
  const float* B         = (const float*)d_in[2];
  const float* C         = (const float*)d_in[3];
  const float* log_step  = (const float*)d_in[4];
  const float* Dv        = (const float*)d_in[5];
  const float* hidden    = (const float*)d_in[6];
  const float* u         = (const float*)d_in[7];
  const unsigned char* start = (const unsigned char*)d_in[8];  // jax bool -> 1 byte

  char* ws = (char*)d_ws;
  float* lbr = (float*)(ws + WS_LB_RE);
  float* lbi = (float*)(ws + WS_LB_IM);
  float* fr  = (float*)(ws + WS_F_RE);
  float* fi  = (float*)(ws + WS_F_IM);
  unsigned short* bbt = (unsigned short*)(ws + WS_BBT);
  unsigned short* c2t = (unsigned short*)(ws + WS_C2T);
  float* aer = (float*)(ws + WS_AER);
  float* aei = (float*)(ws + WS_AEI);
  float* cer = (float*)(ws + WS_CER);
  float* cei = (float*)(ws + WS_CEI);
  float* xr  = (float*)(ws + WS_XR);
  float* xi  = (float*)(ws + WS_XI);
  float* bu  = (float*)(ws + WS_BU);

  float* out_final = (float*)d_out;        // (1,1,P) complex64 = 512 f32
  float* ys        = (float*)d_out + 512;  // (L,H) f32

  prep_modes<<<1, 256, 0, stream>>>(Lambda_re, Lambda_im, log_step, lbr, lbi, fr, fi);
  prep_bbar<<<P_DIM, H_DIM, 0, stream>>>(B, fr, fi, bbt);
  prep_c2t<<<H_DIM, P_DIM, 0, stream>>>(C, c2t);

  gemm_bu<<<dim3(L_SEQ / 128, 512 / 64), 256, 0, stream>>>(u, bbt, bu);

  scan_chunk_reduce<<<NCH, P_DIM, 0, stream>>>(bu, start, lbr, lbi, aer, aei, cer, cei);
  scan_carry<<<1, P_DIM, 0, stream>>>(aer, aei, cer, cei, hidden, xr, xi);
  scan_apply<<<NCH, P_DIM, 0, stream>>>(bu, start, lbr, lbi, xr, xi, out_final);

  gemm_ys<<<dim3(L_SEQ / 128, H_DIM / 64), 256, 0, stream>>>(bu, c2t, u, Dv, ys);
}
